// InfoNCELoss_44547400794504
// MI455X (gfx1250) — matrix operand order repaired and checked by reading, not compiled, not run on hardware
//
#include <hip/hip_runtime.h>
#include <math.h>

typedef __attribute__((ext_vector_type(16))) __bf16 v16bf;
typedef __attribute__((ext_vector_type(8)))  __bf16 v8bf;
typedef __attribute__((ext_vector_type(8)))  float  v8f;

#define LOG2E     1.4426950408889634f
#define LSE_SHIFT 64.0f
#define SHIFT_L2  92.33248261689386f   /* 64 * log2(e) */

// Raw v_exp_f32: 2^x with hardware flush-to-zero on deep underflow.
// Exactly the semantics the fixed-shift LSE wants (tiny terms -> 0).
#if defined(__has_builtin) && __has_builtin(__builtin_amdgcn_exp2f)
#define EXP2_FAST(x) __builtin_amdgcn_exp2f(x)
#else
#define EXP2_FAST(x) exp2f(x)
#endif

__device__ __forceinline__ unsigned f2bf(float x) {
    unsigned u = __float_as_uint(x);
    unsigned r = u + 0x7FFFu + ((u >> 16) & 1u);   // round-to-nearest-even
    return r >> 16;
}

// ---------------------------------------------------------------------------
// fp32 -> bf16 conversion (8 floats / thread), packed 2 halves per dword
// ---------------------------------------------------------------------------
__global__ void cvt_bf16_kernel(const float4* __restrict__ in,
                                uint4* __restrict__ out, int n8) {
    int i = blockIdx.x * blockDim.x + threadIdx.x;
    if (i >= n8) return;
    float4 f0 = in[2 * i];
    float4 f1 = in[2 * i + 1];
    uint4 o;
    o.x = f2bf(f0.x) | (f2bf(f0.y) << 16);
    o.y = f2bf(f0.z) | (f2bf(f0.w) << 16);
    o.z = f2bf(f1.x) | (f2bf(f1.y) << 16);
    o.w = f2bf(f1.z) | (f2bf(f1.w) << 16);
    out[i] = o;
}

// ---------------------------------------------------------------------------
// Streaming GEMM + fixed-shift exp-sum.
//   A: [N][256] bf16 row-major (rows whose LSE we compute)
//   B: [N][256] bf16 row-major (the "columns": logits = A . B^T)
// Workgroup = 8 waves; each wave owns 2 row-tiles of 16 (256 rows / WG).
// grid.y = 4 column slices; partial sums written to sumPart[slice*N + row].
// ---------------------------------------------------------------------------
__global__ __launch_bounds__(256)
void gemm_lse_kernel(const v8bf* __restrict__ A8, const v8bf* __restrict__ B8,
                     float* __restrict__ sumPart, int N) {
    const int lane    = threadIdx.x & 31;
    const int wave    = threadIdx.x >> 5;
    const int laneRow = lane & 15;
    const int sub     = lane >> 4;      // which 16-lane half

    // --- A fragments: 2 row tiles x 8 K-chunks (K=256), kept in VGPRs ------
    // 16-bit A 16x32 layout: lanes 0-15 hold K {0..7, 16..23}, lanes 16-31
    // hold K {8..15, 24..31}  (ISA 7.12.2)
    v16bf a[2][8];
#pragma unroll
    for (int t = 0; t < 2; ++t) {
        int row  = blockIdx.x * 256 + wave * 32 + t * 16 + laneRow;
        int base = row * 32;                       // row stride = 256 halfs = 32 v8bf
#pragma unroll
        for (int c = 0; c < 8; ++c) {
            v8bf lo = A8[base + c * 4 + sub];      // K = 32c + 8*sub + [0..7]
            v8bf hi = A8[base + c * 4 + 2 + sub];  // K = 32c + 16 + 8*sub + [0..7]
            a[t][c] = __builtin_shufflevector(lo, hi,
                        0,1,2,3,4,5,6,7,8,9,10,11,12,13,14,15);
        }
    }

    float s[2][8];
#pragma unroll
    for (int t = 0; t < 2; ++t)
#pragma unroll
        for (int r = 0; r < 8; ++r) s[t][r] = 0.0f;

    const int tilesPerSlice = (N / 16) / 4;
    const int tile0 = blockIdx.y * tilesPerSlice;

    for (int it = 0; it < tilesPerSlice; ++it) {
        // B fragment: 16-bit B 32x16 layout: lanes 0-15 hold K {0..7, 16..23} of
        // column n=lane, lanes 16-31 hold K {8..15, 24..31} (matches A).
        int col   = (tile0 + it) * 16 + laneRow;
        int cbase = col * 32 + sub;

        v8f acc0, acc1;
#pragma unroll
        for (int k = 0; k < 8; ++k) { acc0[k] = 0.0f; acc1[k] = 0.0f; }

#pragma unroll
        for (int c = 0; c < 8; ++c) {
            v8bf blo = B8[cbase + c * 4];
            v8bf bhi = B8[cbase + c * 4 + 2];
            v16bf b = __builtin_shufflevector(blo, bhi,
                        0,1,2,3,4,5,6,7,8,9,10,11,12,13,14,15);
            acc0 = __builtin_amdgcn_wmma_f32_16x16x32_bf16(
                       false, a[0][c], false, b, (short)0, acc0, false, false);
            acc1 = __builtin_amdgcn_wmma_f32_16x16x32_bf16(
                       false, a[1][c], false, b, (short)0, acc1, false, false);
        }

        // fixed-shift exp accumulation: s += 2^(v*log2e - 64*log2e)
#pragma unroll
        for (int r = 0; r < 8; ++r) {
            s[0][r] += EXP2_FAST(fmaf(acc0[r], LOG2E, -SHIFT_L2));
            s[1][r] += EXP2_FAST(fmaf(acc1[r], LOG2E, -SHIFT_L2));
        }
        __syncthreads();   // keep waves in lockstep for WGP$ reuse of B tile
    }

    // Row r (C layout): lanes 0-15 hold M=r, lanes 16-31 hold M=r+8.
    // xor-reduce within each 16-lane half sums the 16 columns per row.
#pragma unroll
    for (int t = 0; t < 2; ++t)
#pragma unroll
        for (int r = 0; r < 8; ++r) {
            float v = s[t][r];
            v += __shfl_xor(v, 1, 32);
            v += __shfl_xor(v, 2, 32);
            v += __shfl_xor(v, 4, 32);
            v += __shfl_xor(v, 8, 32);
            s[t][r] = v;
        }

    if (laneRow == 0) {
#pragma unroll
        for (int t = 0; t < 2; ++t)
#pragma unroll
            for (int r = 0; r < 8; ++r) {
                int row = blockIdx.x * 256 + wave * 32 + t * 16 + sub * 8 + r;
                sumPart[blockIdx.y * N + row] = s[t][r];
            }
    }
}

// ---------------------------------------------------------------------------
// Per-row loss: fp32-exact diagonal dot + LSE assembly; one wave per row.
// blockPart[blockIdx] = sum of 8 rows' contributions.
// ---------------------------------------------------------------------------
__global__ __launch_bounds__(256)
void loss_partial_kernel(const float* __restrict__ img,
                         const float* __restrict__ txt,
                         const float* __restrict__ sumRow,
                         const float* __restrict__ sumCol,
                         float* __restrict__ blockPart, int N) {
    __shared__ float lds[8];
    const int lane = threadIdx.x & 31;
    const int wave = threadIdx.x >> 5;
    const int row  = blockIdx.x * 8 + wave;

    const float4* i4 = (const float4*)(img + (size_t)row * 256);
    const float4* t4 = (const float4*)(txt + (size_t)row * 256);
    float4 a0 = i4[lane * 2], a1 = i4[lane * 2 + 1];
    float4 b0 = t4[lane * 2], b1 = t4[lane * 2 + 1];
    float dot = 0.0f;
    dot = fmaf(a0.x, b0.x, dot); dot = fmaf(a0.y, b0.y, dot);
    dot = fmaf(a0.z, b0.z, dot); dot = fmaf(a0.w, b0.w, dot);
    dot = fmaf(a1.x, b1.x, dot); dot = fmaf(a1.y, b1.y, dot);
    dot = fmaf(a1.z, b1.z, dot); dot = fmaf(a1.w, b1.w, dot);
    dot += __shfl_xor(dot, 16, 32);
    dot += __shfl_xor(dot, 8, 32);
    dot += __shfl_xor(dot, 4, 32);
    dot += __shfl_xor(dot, 2, 32);
    dot += __shfl_xor(dot, 1, 32);

    if (lane == 0) {
        float sr = sumRow[row] + sumRow[N + row] + sumRow[2 * N + row] + sumRow[3 * N + row];
        float sc = sumCol[row] + sumCol[N + row] + sumCol[2 * N + row] + sumCol[3 * N + row];
        float lse_r = LSE_SHIFT + logf(sr);
        float lse_c = LSE_SHIFT + logf(sc);
        lds[wave] = 0.5f * (lse_r + lse_c) - dot;
    }
    __syncthreads();
    if (threadIdx.x == 0) {
        float t = 0.0f;
#pragma unroll
        for (int w = 0; w < 8; ++w) t += lds[w];
        blockPart[blockIdx.x] = t;
    }
}

__global__ __launch_bounds__(256)
void final_reduce_kernel(const float* __restrict__ blockPart, float* __restrict__ out,
                         int nPart, float invN) {
    __shared__ float lds[256];
    float s = 0.0f;
    for (int i = threadIdx.x; i < nPart; i += 256) s += blockPart[i];
    lds[threadIdx.x] = s;
    __syncthreads();
    for (int off = 128; off > 0; off >>= 1) {
        if ((int)threadIdx.x < off) lds[threadIdx.x] += lds[threadIdx.x + off];
        __syncthreads();
    }
    if (threadIdx.x == 0) out[0] = lds[0] * invN;
}

// ---------------------------------------------------------------------------
extern "C" void kernel_launch(void* const* d_in, const int* in_sizes, int n_in,
                              void* d_out, int out_size, void* d_ws, size_t ws_size,
                              hipStream_t stream) {
    (void)n_in; (void)out_size; (void)ws_size;
    const float* img = (const float*)d_in[0];
    const float* txt = (const float*)d_in[1];
    float* out = (float*)d_out;

    const int D = 256;
    const int N = in_sizes[0] / D;          // 16384

    char* ws = (char*)d_ws;
    size_t bfBytes = (size_t)N * D * 2;     // 8 MB each
    void*  imgBf   = (void*)ws;
    void*  txtBf   = (void*)(ws + bfBytes);
    float* sumRow  = (float*)(ws + 2 * bfBytes);       // [4][N]
    float* sumCol  = sumRow + 4 * (size_t)N;           // [4][N]
    float* blockPart = sumCol + 4 * (size_t)N;         // [N/8]

    int n8 = N * D / 8;
    cvt_bf16_kernel<<<n8 / 256, 256, 0, stream>>>((const float4*)img, (uint4*)imgBf, n8);
    cvt_bf16_kernel<<<n8 / 256, 256, 0, stream>>>((const float4*)txt, (uint4*)txtBf, n8);

    dim3 g(N / 256, 4);
    gemm_lse_kernel<<<g, 256, 0, stream>>>((const v8bf*)imgBf, (const v8bf*)txtBf, sumRow, N);
    gemm_lse_kernel<<<g, 256, 0, stream>>>((const v8bf*)txtBf, (const v8bf*)imgBf, sumCol, N);

    loss_partial_kernel<<<N / 8, 256, 0, stream>>>(img, txt, sumRow, sumCol, blockPart, N);
    final_reduce_kernel<<<1, 256, 0, stream>>>(blockPart, out, N / 8, 1.0f / (float)N);
}
